// SCARAFE_70239895159131
// MI455X (gfx1250) — matrix operand order repaired and checked by reading, not compiled, not run on hardware
//
#include <hip/hip_runtime.h>
#include <hip/hip_fp16.h>

// CDNA5 / gfx1250 WMMA fragment types (wave32)
typedef __attribute__((ext_vector_type(16))) _Float16 v16h;
typedef __attribute__((ext_vector_type(8)))  _Float16 v8h;
typedef __attribute__((ext_vector_type(8)))  float    v8f;
typedef __attribute__((ext_vector_type(4)))  int      v4i;

// address-space-qualified vector pointers for the async-to-LDS builtin
typedef __attribute__((address_space(1))) v4i gas_v4i;   // global
typedef __attribute__((address_space(3))) v4i las_v4i;   // LDS

#define BN_EPS 1e-5f

#if __has_builtin(__builtin_amdgcn_global_load_async_to_lds_b128) && \
    __has_builtin(__builtin_amdgcn_s_wait_asynccnt)
#define USE_ASYNC_LDS 1
#else
#define USE_ASYNC_LDS 0
#endif

__device__ __forceinline__ float silu_f(float y) {
    return y / (1.0f + __expf(-y));
}

// A-fragment (16x32 f16, MxK) from f32 weights: lane holds row M=lane%16.
// elem e -> K = k0 + (e/8)*16 + (lane/16)*8 + (e%8).
__device__ __forceinline__ void fill_a_f32(v16h& a, const float* __restrict__ ap) {
    float4 A0 = *(const float4*)(ap);
    float4 A1 = *(const float4*)(ap + 4);
    float4 A2 = *(const float4*)(ap + 16);
    float4 A3 = *(const float4*)(ap + 20);
    a[0]=(_Float16)A0.x;  a[1]=(_Float16)A0.y;  a[2]=(_Float16)A0.z;  a[3]=(_Float16)A0.w;
    a[4]=(_Float16)A1.x;  a[5]=(_Float16)A1.y;  a[6]=(_Float16)A1.z;  a[7]=(_Float16)A1.w;
    a[8]=(_Float16)A2.x;  a[9]=(_Float16)A2.y;  a[10]=(_Float16)A2.z; a[11]=(_Float16)A2.w;
    a[12]=(_Float16)A3.x; a[13]=(_Float16)A3.y; a[14]=(_Float16)A3.z; a[15]=(_Float16)A3.w;
}

// Same A layout, directly from f16 memory (two contiguous 8-half runs, 16 apart).
__device__ __forceinline__ void fill_a_f16(v16h& a, const _Float16* __restrict__ ap) {
    v8h lo = *(const v8h*)ap;
    v8h hi = *(const v8h*)(ap + 16);
#pragma unroll
    for (int e = 0; e < 8; ++e) { a[e] = lo[e]; a[e + 8] = hi[e]; }
}

// B-fragment (32x16 f16, KxN): lane holds col N=lane%16, elem e -> K=(e/8)*16+(lane/16)*8+(e%8).
__device__ __forceinline__ v16h load_b_lds(const _Float16* __restrict__ bp) {
    v8h lo = *(const v8h*)bp;
    v8h hi = *(const v8h*)(bp + 16);
    v16h b;
#pragma unroll
    for (int e = 0; e < 8; ++e) { b[e] = lo[e]; b[e + 8] = hi[e]; }
    return b;
}

// ---------------------------------------------------------------------------
// K0: one-shot repack of w2 (OIHW f32, stride-9 inner) -> tap-major f16
// w2t[tap][m*128+ci].  Makes K2's A-operand two 16B loads per WMMA.
// ---------------------------------------------------------------------------
__global__ __launch_bounds__(256) void k0_prep_w2(
    const float* __restrict__ w2, _Float16* __restrict__ w2t)
{
    int idx = blockIdx.x * 256 + threadIdx.x;   // 147456 = 576*256 exactly
    int tap = idx % 9;
    int mc  = idx / 9;                          // m*128 + ci
    w2t[(size_t)tap * 16384 + mc] = (_Float16)w2[(size_t)mc * 9 + tap];
}

// ---------------------------------------------------------------------------
// K1: 1x1 conv 256->128 + BN + SiLU.  GEMM M=128, K=256, N=16 px per block.
// ---------------------------------------------------------------------------
__global__ __launch_bounds__(256) void k1_reduce(
    const float* __restrict__ x, const float* __restrict__ w1,
    const float* __restrict__ g1, const float* __restrict__ b1,
    const float* __restrict__ m1, const float* __restrict__ v1,
    _Float16* __restrict__ f_ws)
{
    __shared__ alignas(32) _Float16 ldsB[16 * 264];   // [n][k], ld=264 (bank-stride 4)
    const int tid = threadIdx.x;
    const int b   = blockIdx.x >> 8;
    const int p0  = (blockIdx.x & 255) << 4;

    {   // stage x[k=tid][p0..p0+15] -> LDS f16
        const float* src = x + ((size_t)b * 256 + tid) * 4096 + p0;
#pragma unroll
        for (int n = 0; n < 16; ++n) ldsB[n * 264 + tid] = (_Float16)src[n];
    }
    __syncthreads();

    const int lane = tid & 31, wave = tid >> 5;
    const int halfsel = lane >> 4, n = lane & 15;
    const int mbase = wave << 4;
    const float* wrow = w1 + (size_t)(mbase + n) * 256;

    v8f acc = {0.f, 0.f, 0.f, 0.f, 0.f, 0.f, 0.f, 0.f};
#pragma unroll 1
    for (int k0 = 0; k0 < 256; k0 += 32) {
        v16h a, bf;
        fill_a_f32(a, wrow + k0 + halfsel * 8);
        bf = load_b_lds(&ldsB[n * 264 + k0 + halfsel * 8]);
        acc = __builtin_amdgcn_wmma_f32_16x16x32_f16(false, a, false, bf,
                                                     (short)0, acc, false, false);
    }
#pragma unroll
    for (int r = 0; r < 8; ++r) {
        int m = mbase + halfsel * 8 + r;
        float inv = g1[m] * rsqrtf(v1[m] + BN_EPS);
        float bet = b1[m] - m1[m] * inv;
        float y = acc[r] * inv + bet;
        f_ws[((size_t)b * 128 + m) * 4096 + p0 + n] = (_Float16)silu_f(y);
    }
}

// ---------------------------------------------------------------------------
// K2: 3x3 conv 128->128 + BN + SiLU, implicit GEMM (K = 9 taps x 128 ci).
// ---------------------------------------------------------------------------
__global__ __launch_bounds__(256) void k2_enc3x3(
    const _Float16* __restrict__ f_ws, const _Float16* __restrict__ w2t,
    const float* __restrict__ g2, const float* __restrict__ b2,
    const float* __restrict__ m2, const float* __restrict__ v2,
    _Float16* __restrict__ h_ws)
{
    __shared__ alignas(32) _Float16 ldsF[54 * 136];   // [(ry*18+rx)][ci], ld=136
    const int tid = threadIdx.x;
    const int b   = blockIdx.x >> 8;
    const int pt  = blockIdx.x & 255;
    const int row = pt >> 2;
    const int c0  = (pt & 3) << 4;
    const _Float16* fb = f_ws + (size_t)b * 128 * 4096;

    for (int idx = tid; idx < 54 * 128; idx += 256) {   // 6912 = 27*256, uniform
        int s = idx % 54, ci = idx / 54;
        int ry = s / 18, rx = s % 18;
        int gr = row + ry - 1, gc = c0 + rx - 1;
        _Float16 val = (_Float16)0.f;
        if ((unsigned)gr < 64u && (unsigned)gc < 64u)
            val = fb[(size_t)ci * 4096 + gr * 64 + gc];
        ldsF[s * 136 + ci] = val;
    }
    __syncthreads();

    const int lane = tid & 31, wave = tid >> 5;
    const int halfsel = lane >> 4, n = lane & 15;
    const int m = (wave << 4) + n;

    v8f acc = {0.f, 0.f, 0.f, 0.f, 0.f, 0.f, 0.f, 0.f};
#pragma unroll 1
    for (int tap = 0; tap < 9; ++tap) {
        int dy = tap / 3, dx = tap % 3;
        int srow = dy * 18 + dx + n;
        const _Float16* arow = w2t + (size_t)tap * 16384 + (size_t)m * 128;
#pragma unroll 1
        for (int k0 = 0; k0 < 128; k0 += 32) {
            v16h a, bf;
            fill_a_f16(a, arow + k0 + halfsel * 8);
            bf = load_b_lds(&ldsF[srow * 136 + k0 + halfsel * 8]);
            acc = __builtin_amdgcn_wmma_f32_16x16x32_f16(false, a, false, bf,
                                                         (short)0, acc, false, false);
        }
    }
    const int p = row * 64 + c0 + n;
#pragma unroll
    for (int r = 0; r < 8; ++r) {
        int mo = (wave << 4) + halfsel * 8 + r;
        float inv = g2[mo] * rsqrtf(v2[mo] + BN_EPS);
        float bet = b2[mo] - m2[mo] * inv;
        float y = acc[r] * inv + bet;
        h_ws[((size_t)b * 128 + mo) * 4096 + p] = (_Float16)silu_f(y);
    }
}

// ---------------------------------------------------------------------------
// K3: 1x1 conv 128->25 + bias + softmax(25). M padded to 32 (2 waves/group),
// 4 groups of 16 px per block. k stored f32, tap-major, coalesced.
// ---------------------------------------------------------------------------
__global__ __launch_bounds__(256) void k3_kernel_softmax(
    const _Float16* __restrict__ h_ws, const float* __restrict__ w3,
    const float* __restrict__ b3, float* __restrict__ k_ws)
{
    __shared__ alignas(32) _Float16 ldsB[4][16 * 136];
    __shared__ float ldsL[4][32 * 16];
    const int tid = threadIdx.x;
    const int b   = blockIdx.x >> 6;
    const int p0  = (blockIdx.x & 63) << 6;
    const _Float16* hb = h_ws + (size_t)b * 128 * 4096 + p0;

    for (int idx = tid; idx < 8192; idx += 256) {
        int q = idx & 63, k = idx >> 6;
        ldsB[q >> 4][(q & 15) * 136 + k] = hb[(size_t)k * 4096 + q];
    }
    __syncthreads();

    const int lane = tid & 31, wave = tid >> 5;
    const int g = wave >> 1, wm = wave & 1;
    const int halfsel = lane >> 4, n = lane & 15;
    const int mg = wm * 16 + n;
    const bool mvalid = mg < 25;
    const float* wrow = w3 + (size_t)mg * 128;

    v8f acc = {0.f, 0.f, 0.f, 0.f, 0.f, 0.f, 0.f, 0.f};
#pragma unroll 1
    for (int k0 = 0; k0 < 128; k0 += 32) {
        v16h a, bf;
        if (mvalid) {
            fill_a_f32(a, wrow + k0 + halfsel * 8);
        } else {
#pragma unroll
            for (int e = 0; e < 16; ++e) a[e] = (_Float16)0.f;
        }
        bf = load_b_lds(&ldsB[g][n * 136 + k0 + halfsel * 8]);
        acc = __builtin_amdgcn_wmma_f32_16x16x32_f16(false, a, false, bf,
                                                     (short)0, acc, false, false);
    }
#pragma unroll
    for (int r = 0; r < 8; ++r) {
        int m = wm * 16 + halfsel * 8 + r;
        float lg = acc[r] + (m < 25 ? b3[m] : 0.f);
        ldsL[g][m * 16 + n] = lg;
    }
    __syncthreads();

    if (tid < 64) {
        int gg = tid >> 4, nn = tid & 15;
        float tmp[25], mx = -1e30f;
#pragma unroll
        for (int m = 0; m < 25; ++m) { tmp[m] = ldsL[gg][m * 16 + nn]; mx = fmaxf(mx, tmp[m]); }
        float sum = 0.f;
#pragma unroll
        for (int m = 0; m < 25; ++m) { tmp[m] = __expf(tmp[m] - mx); sum += tmp[m]; }
        float rs = 1.f / sum;
        int p = p0 + tid;
#pragma unroll
        for (int m = 0; m < 25; ++m)
            k_ws[((size_t)b * 25 + m) * 4096 + p] = tmp[m] * rs;
    }
}

// ---------------------------------------------------------------------------
// K4: content-aware 5x5 gather + nearest 2x upsample (fused).
// The kernel-row staging is a pure aligned copy -> CDNA5 async-to-LDS
// (GLOBAL_LOAD_ASYNC_TO_LDS_B128, ASYNCcnt) when the builtin is available.
// ---------------------------------------------------------------------------
__global__ __launch_bounds__(256) void k4_gather_up(
    const float* __restrict__ x, const float* __restrict__ k_ws,
    float* __restrict__ out)
{
    __shared__ alignas(16) float ldsK[25 * 64];    // [tap][w]
    __shared__ alignas(16) float ldsX[4 * 5 * 68]; // [c][ry][rx]
    const int h  = blockIdx.x;
    const int cb = blockIdx.y;
    const int b  = blockIdx.z;
    const int tid = threadIdx.x;

    const float* xb = x + (size_t)(b * 256 + cb * 4) * 4096;
    const float* kb = k_ws + (size_t)b * 25 * 4096 + h * 64;

#if USE_ASYNC_LDS
    // 1600 floats = 400 x 16B chunks; 64-float rows are 16B aligned and
    // chunks never cross a tap row.  Async DMA global -> LDS (ASYNCcnt).
    for (int c4 = tid; c4 < 400; c4 += 256) {
        int fi = c4 * 4;
        const float* gsrc = kb + (size_t)(fi >> 6) * 4096 + (fi & 63);
        __builtin_amdgcn_global_load_async_to_lds_b128(
            (gas_v4i*)(void*)gsrc,
            (las_v4i*)(void*)&ldsK[fi],
            0, 0);
    }
#else
    for (int idx = tid; idx < 1600; idx += 256)
        ldsK[idx] = kb[(size_t)(idx >> 6) * 4096 + (idx & 63)];
#endif
    for (int idx = tid; idx < 1360; idx += 256) {
        int c = idx / 340, r2 = idx % 340;
        int ry = r2 / 68, rx = r2 % 68;
        int gh = h + ry - 2, gw = rx - 2;
        float v = 0.f;
        if ((unsigned)gh < 64u && (unsigned)gw < 64u)
            v = xb[(size_t)c * 4096 + gh * 64 + gw];
        ldsX[idx] = v;
    }
#if USE_ASYNC_LDS
    __builtin_amdgcn_s_wait_asynccnt(0);   // each wave drains its own DMAs
#endif
    __syncthreads();

    const int c = tid >> 6, w = tid & 63;
    float s = 0.f;
#pragma unroll
    for (int i = 0; i < 5; ++i)
#pragma unroll
        for (int j = 0; j < 5; ++j)
            s = fmaf(ldsX[c * 340 + i * 68 + w + j], ldsK[(i * 5 + j) * 64 + w], s);

    float2 vv = make_float2(s, s);
    size_t obase = (((size_t)(b * 256 + cb * 4 + c)) * 128 + 2 * h) * 128 + 2 * w;
    *(float2*)(out + obase)       = vv;   // row 2h
    *(float2*)(out + obase + 128) = vv;   // row 2h+1
}

// ---------------------------------------------------------------------------
extern "C" void kernel_launch(void* const* d_in, const int* in_sizes, int n_in,
                              void* d_out, int out_size, void* d_ws, size_t ws_size,
                              hipStream_t stream)
{
    (void)in_sizes; (void)n_in; (void)out_size; (void)ws_size;
    const float* x  = (const float*)d_in[0];
    const float* w1 = (const float*)d_in[1];
    const float* g1 = (const float*)d_in[2];
    const float* b1 = (const float*)d_in[3];
    const float* m1 = (const float*)d_in[4];
    const float* v1 = (const float*)d_in[5];
    const float* w2 = (const float*)d_in[6];
    const float* g2 = (const float*)d_in[7];
    const float* b2 = (const float*)d_in[8];
    const float* m2 = (const float*)d_in[9];
    const float* v2 = (const float*)d_in[10];
    const float* w3 = (const float*)d_in[11];
    const float* b3 = (const float*)d_in[12];
    float* out = (float*)d_out;

    char* ws = (char*)d_ws;
    _Float16* f_ws  = (_Float16*)(ws);                   // 8 MB  (8*128*4096 f16)
    _Float16* h_ws  = (_Float16*)(ws + (8u  << 20));     // 8 MB
    float*    k_ws  = (float*)   (ws + (16u << 20));     // 3.2 MB (8*25*4096 f32)
    _Float16* w2t   = (_Float16*)(ws + (20u << 20));     // 288 KB (9*128*128 f16)

    k0_prep_w2       <<<576,  256, 0, stream>>>(w2, w2t);
    k1_reduce        <<<2048, 256, 0, stream>>>(x, w1, g1, b1, m1, v1, f_ws);
    k2_enc3x3        <<<2048, 256, 0, stream>>>(f_ws, w2t, g2, b2, m2, v2, h_ws);
    k3_kernel_softmax<<<512,  256, 0, stream>>>(h_ws, w3, b3, k_ws);
    k4_gather_up     <<<dim3(64, 64, 8), 256, 0, stream>>>(x, k_ws, out);
}
